// Ouroboros6Model_68685116998073
// MI455X (gfx1250) — matrix operand order repaired and checked by reading, not compiled, not run on hardware
//
#include <hip/hip_runtime.h>

// ---------------- model constants ----------------
constexpr int D      = 320;
constexpr int NH     = 5;
constexpr int HD     = 64;
constexpr int FF     = 864;
constexpr int NL     = 6;
constexpr int NLOOPS = 8;
constexpr int B      = 8;
constexpr int S      = 2048;
constexpr int T      = B * S;        // 16384 tokens
constexpr int KSEL   = 1638;         // int(S * 0.8)

typedef _Float16 v16h __attribute__((ext_vector_type(16)));
typedef _Float16 v8h  __attribute__((ext_vector_type(8)));
typedef float    v8f  __attribute__((ext_vector_type(8)));

// ---------------- WMMA helpers (CDNA5, wave32) ----------------
__device__ inline v8f wmma16(v16h a, v16h b, v8f c) {
  // D = A(16x32 f16) * B(32x16 f16) + C(16x16 f32)
  return __builtin_amdgcn_wmma_f32_16x16x32_f16(false, a, false, b, (short)0, c,
                                                false, false);
}

// A fragment from f16 LDS tile [rows x ld], row m = lane&15,
// K = kb..kb+7 and kb+16..kb+23 with kb = 8*(lane>>4)  -> two 16B LDS loads
__device__ inline v16h frag_a_h(const _Float16* p, int row0, int ld) {
  const int lane = threadIdx.x & 31;
  const int m  = lane & 15;
  const int kb = (lane >> 4) * 8;
  const _Float16* r = p + (size_t)(row0 + m) * ld + kb;
  v8h lo = *(const v8h*)(r);
  v8h hi = *(const v8h*)(r + 16);
  return __builtin_shufflevector(lo, hi, 0, 1, 2, 3, 4, 5, 6, 7,
                                 8, 9, 10, 11, 12, 13, 14, 15);
}

// B fragment from f16 LDS tile stored [N rows x K cols] row-major:
// lane n = lane&15, K = kb..kb+7 and kb+16..kb+23 with kb = 8*(lane>>4)  -> two 16B LDS loads
__device__ inline v16h frag_b_h(const _Float16* p, int row0, int ld) {
  const int lane = threadIdx.x & 31;
  const int n  = lane & 15;
  const int kb = (lane >> 4) * 8;
  const _Float16* r = p + (size_t)(row0 + n) * ld + kb;
  v8h lo = *(const v8h*)(r);
  v8h hi = *(const v8h*)(r + 16);
  return __builtin_shufflevector(lo, hi, 0, 1, 2, 3, 4, 5, 6, 7,
                                 8, 9, 10, 11, 12, 13, 14, 15);
}

// ---------------- CDNA5 async global->LDS copy (ASYNCcnt tracked) -----------
__device__ inline void async_ld16(unsigned int lds_off, unsigned long long gaddr) {
  asm volatile("global_load_async_to_lds_b128 %0, %1, off"
               :: "v"(lds_off), "v"(gaddr) : "memory");
}
__device__ inline void wait_async0() {
  asm volatile("s_wait_asynccnt 0x0" ::: "memory");
}

// ---------------- embedding + iteration embedding ----------------
__global__ void embed_kernel(const int* __restrict__ ids, const int* __restrict__ iter,
                             const float* __restrict__ emb, const float* __restrict__ iemb,
                             float* __restrict__ x) {
  int i = blockIdx.x * blockDim.x + threadIdx.x;
  if (i >= T * D) return;
  int t = i / D, d = i - t * D;
  float v = emb[(size_t)ids[t] * D + d];
  int it = iter[0];
  if (it < NLOOPS) v += iemb[it * D + d];
  x[i] = v;
}

// ---------------- RMSNorm (one wave per row, optional gather) ----------------
__global__ __launch_bounds__(256) void rmsnorm_kernel(
    const float* __restrict__ x, const float* __restrict__ w,
    float* __restrict__ y, int rows, const int* __restrict__ rowmap) {
  int row = blockIdx.x * 8 + (threadIdx.x >> 5);
  if (row >= rows) return;
  int lane = threadIdx.x & 31;
  int src = rowmap ? rowmap[row] : row;
  const float* xi = x + (size_t)src * D;
  float s = 0.f;
  for (int d = lane; d < D; d += 32) { float v = xi[d]; s += v * v; }
#pragma unroll
  for (int o = 16; o > 0; o >>= 1) s += __shfl_xor(s, o);
  float inv = rsqrtf(s / (float)D + 1e-6f);
  float* yo = y + (size_t)row * D;
  for (int d = lane; d < D; d += 32) yo[d] = w[d] * xi[d] * inv;
}

// ---------------- generic WMMA GEMM: C = epi(A[MxK] * W[NxK]^T) ----------------
// 8 waves, block tile 64(M) x 128(N), K-step 32. Tiles converted f32->f16 once
// at staging; fragments are pure ds_load_b128 (no per-WMMA cvts).
// epi: 0=store 1=add-in-place 2=store silu 3=multiply-existing 4=scatter-add*scale
__global__ __launch_bounds__(256) void gemm_kernel(
    const float* __restrict__ A, int lda,
    const float* __restrict__ W, int ldw,
    float* __restrict__ C, int ldc,
    int M, int N, int K, int epi,
    const int* __restrict__ rowmap, const float* __restrict__ rowscale) {
  __shared__ _Float16 sA[64 * 32];
  __shared__ _Float16 sW[128 * 32];
  const int tid = threadIdx.x;
  const int lane = tid & 31;
  const int wv = tid >> 5;
  const int wr = wv >> 1;                 // wave row 0..3  (16 rows each)
  const int wc = wv & 1;                  // wave col 0..1  (64 cols each)
  const int m0 = blockIdx.y * 64;
  const int n0 = blockIdx.x * 128;

  // per-thread staging chunks (contiguous within one tile row)
  const int aBase = tid * 8;              // A: 2048 elems, 8 per thread
  const int aRow = aBase >> 5, aCol = aBase & 31;
  const int wBase = tid * 16;             // W: 4096 elems, 16 per thread
  const int wRow = wBase >> 5, wCol = wBase & 31;

  v8f acc[4] = {};
  for (int k0 = 0; k0 < K; k0 += 32) {
    __syncthreads();
    {
      v8h hv;
      int gm = m0 + aRow;
      if (gm < M) {
        const float* src = A + (size_t)gm * lda + k0 + aCol;
#pragma unroll
        for (int e = 0; e < 8; ++e) hv[e] = (_Float16)src[e];
      } else {
#pragma unroll
        for (int e = 0; e < 8; ++e) hv[e] = (_Float16)0.f;
      }
      *(v8h*)(sA + aBase) = hv;
    }
    {
      v16h hv;
      int gn = n0 + wRow;
      if (gn < N) {
        const float* src = W + (size_t)gn * ldw + k0 + wCol;
#pragma unroll
        for (int e = 0; e < 16; ++e) hv[e] = (_Float16)src[e];
      } else {
#pragma unroll
        for (int e = 0; e < 16; ++e) hv[e] = (_Float16)0.f;
      }
      *(v16h*)(sW + wBase) = hv;
    }
    if (k0 + 32 < K) {                    // prefetch next K tile into caches
      int r = tid >> 1;
      if (m0 + (r & 63) < M) __builtin_prefetch(&A[(size_t)(m0 + (r & 63)) * lda + k0 + 32], 0, 1);
      if (n0 + r < N)        __builtin_prefetch(&W[(size_t)(n0 + r) * ldw + k0 + 32], 0, 1);
    }
    __syncthreads();
    v16h af = frag_a_h(sA, wr * 16, 32);
#pragma unroll
    for (int t = 0; t < 4; ++t) {
      v16h bf = frag_b_h(sW, wc * 64 + t * 16, 32);
      acc[t] = wmma16(af, bf, acc[t]);
    }
  }
  const int hi = lane >> 4, nn = lane & 15;
#pragma unroll
  for (int t = 0; t < 4; ++t) {
#pragma unroll
    for (int r = 0; r < 8; ++r) {
      int m = m0 + wr * 16 + r + 8 * hi;
      int n = n0 + wc * 64 + t * 16 + nn;
      if (m >= M || n >= N) continue;
      float v = acc[t][r];
      if (epi == 0) {
        C[(size_t)m * ldc + n] = v;
      } else if (epi == 1) {
        C[(size_t)m * ldc + n] += v;
      } else if (epi == 2) {
        C[(size_t)m * ldc + n] = v / (1.0f + __expf(-v));   // silu
      } else if (epi == 3) {
        C[(size_t)m * ldc + n] *= v;
      } else {
        int row = rowmap[m];
        C[(size_t)row * ldc + n] += v * rowscale[m];
      }
    }
  }
}

// ---------------- RoPE: qkv f32 -> f16 q (pre-scaled), k, v-transposed --------
// q,k: [B,NH,S,HD] f16 ; vt: [B,NH,HD,S] f16 (so V^T tiles are row-contiguous)
__global__ void rope_kernel(const float* __restrict__ qkv,
                            _Float16* __restrict__ qh, _Float16* __restrict__ kh,
                            _Float16* __restrict__ vt) {
  int i = blockIdx.x * blockDim.x + threadIdx.x;
  if (i >= T * NH * HD) return;
  int hd = i % HD;
  int h  = (i / HD) % NH;
  int t  = i / (HD * NH);
  int s  = t % S, b = t / S;
  const float* base = qkv + (size_t)t * 3 * D;
  float qv = base[(0 * NH + h) * HD + hd];
  float kv = base[(1 * NH + h) * HD + hd];
  float vv = base[(2 * NH + h) * HD + hd];
  int half = hd & 31;
  float freq = __powf(10000.0f, -(float)half / 32.0f);
  float ang = (float)s * freq;
  float sn, c;
  __sincosf(ang, &sn, &c);
  float qr = (hd < 32) ? -base[(0 * NH + h) * HD + hd + 32]
                       :  base[(0 * NH + h) * HD + hd - 32];
  float kr = (hd < 32) ? -base[(1 * NH + h) * HD + hd + 32]
                       :  base[(1 * NH + h) * HD + hd - 32];
  size_t o = (((size_t)b * NH + h) * S + s) * HD + hd;
  qh[o] = (_Float16)((qv * c + qr * sn) * 0.125f);   // fold 1/sqrt(HD)
  kh[o] = (_Float16)(kv * c + kr * sn);
  vt[(((size_t)b * NH + h) * HD + hd) * S + s] = (_Float16)vv;
}

// ---------------- flash attention (causal), 128 queries/block, 32-key tiles ----
// K/V^T tiles stream into LDS via async global->LDS b128 copies (ASYNCcnt).
// Output written in [B,S,NH,HD] layout (== [T,D] for the Wo GEMM).
__global__ __launch_bounds__(256) void attn_kernel(
    const _Float16* __restrict__ Q, const _Float16* __restrict__ K,
    const _Float16* __restrict__ V, float* __restrict__ O) {
  __shared__ _Float16 sKh[32 * 64];       // [key][hd]     4KB
  __shared__ _Float16 sVt[64 * 32];       // [hd][key]     4KB
  __shared__ _Float16 sPh[8 * 16 * 32];   // per-wave P    8KB
  const int tid = threadIdx.x, lane = tid & 31, wv = tid >> 5;
  const int bh = blockIdx.y;
  const int b = bh / NH, h = bh - b * NH;
  const int q0 = blockIdx.x * 128 + wv * 16; // this wave's 16 query rows
  const _Float16* Qb = Q + (size_t)bh * S * HD;
  const _Float16* Kb = K + (size_t)bh * S * HD;
  const _Float16* Vb = V + (size_t)bh * HD * S;
  const int m = lane & 15, hi = lane >> 4;
  const int kb8 = hi * 8;

  // Q fragments straight from global f16 (scale already folded in)
  v16h qa0, qa1;
  {
    const _Float16* qr = Qb + (size_t)(q0 + m) * HD;
    v8h a0 = *(const v8h*)(qr + kb8);
    v8h a1 = *(const v8h*)(qr + 16 + kb8);
    v8h a2 = *(const v8h*)(qr + 32 + kb8);
    v8h a3 = *(const v8h*)(qr + 48 + kb8);
    qa0 = __builtin_shufflevector(a0, a1, 0,1,2,3,4,5,6,7,8,9,10,11,12,13,14,15);
    qa1 = __builtin_shufflevector(a2, a3, 0,1,2,3,4,5,6,7,8,9,10,11,12,13,14,15);
  }

  const unsigned int sK_off = (unsigned int)(uintptr_t)sKh;
  const unsigned int sV_off = (unsigned int)(uintptr_t)sVt;

  float mrow[8], lrow[8];
#pragma unroll
  for (int r = 0; r < 8; ++r) { mrow[r] = -1e30f; lrow[r] = 0.f; }
  v8f oacc[4] = {};

  const int ktiles = (blockIdx.x + 1) * 4;   // keys up to block's last query
  for (int jt = 0; jt < ktiles; ++jt) {
    const int k0 = jt * 32;
    __syncthreads();
    // K tile: 32x64 f16 = 4096B contiguous; 256 threads x 16B
    async_ld16(sK_off + tid * 16,
               (unsigned long long)(uintptr_t)((const char*)(Kb + (size_t)k0 * HD) + tid * 16));
    // V^T tile: 64 rows x 64B (row stride S*2B); chunk tid -> row tid>>2
    async_ld16(sV_off + tid * 16,
               (unsigned long long)(uintptr_t)((const char*)(Vb + (size_t)(tid >> 2) * S + k0) + (tid & 3) * 16));
    wait_async0();
    __syncthreads();

    // scores: S = Q * K^T  (two 16-col halves, K accumulated over HD=64)
    v8f sc0 = {}, sc1 = {};
    sc0 = wmma16(qa0, frag_b_h(sKh,      0,  64), sc0);
    sc0 = wmma16(qa1, frag_b_h(sKh + 32, 0,  64), sc0);
    sc1 = wmma16(qa0, frag_b_h(sKh,      16, 64), sc1);
    sc1 = wmma16(qa1, frag_b_h(sKh + 32, 16, 64), sc1);

    // causal mask + online softmax (row lives across 16 lanes of one half-wave)
    _Float16* pb = sPh + wv * 16 * 32;
#pragma unroll
    for (int r = 0; r < 8; ++r) {
      int qi = q0 + r + 8 * hi;
      float s0 = (k0 + m      <= qi) ? sc0[r] : -1e30f;
      float s1 = (k0 + 16 + m <= qi) ? sc1[r] : -1e30f;
      float mx = fmaxf(s0, s1);
#pragma unroll
      for (int off = 8; off > 0; off >>= 1) mx = fmaxf(mx, __shfl_xor(mx, off));
      float mnew = fmaxf(mrow[r], mx);
      float corr = __expf(mrow[r] - mnew);
      float p0 = __expf(s0 - mnew);
      float p1 = __expf(s1 - mnew);
      float rs = p0 + p1;
#pragma unroll
      for (int off = 8; off > 0; off >>= 1) rs += __shfl_xor(rs, off);
      lrow[r] = lrow[r] * corr + rs;
      mrow[r] = mnew;
#pragma unroll
      for (int c = 0; c < 4; ++c) oacc[c][r] *= corr;
      pb[(r + 8 * hi) * 32 + m]      = (_Float16)p0;
      pb[(r + 8 * hi) * 32 + 16 + m] = (_Float16)p1;
    }
    __syncthreads();

    // O += P(16x32) * V(32x64)   (V^T tile rows are the B-matrix N rows)
    v16h pf = frag_a_h(pb, 0, 32);
#pragma unroll
    for (int c = 0; c < 4; ++c)
      oacc[c] = wmma16(pf, frag_b_h(sVt, c * 16, 32), oacc[c]);
  }

  // normalize and write [B,S,NH,HD]
#pragma unroll
  for (int c = 0; c < 4; ++c) {
#pragma unroll
    for (int r = 0; r < 8; ++r) {
      int qi = q0 + r + 8 * hi;
      int hd = c * 16 + m;
      O[(((size_t)b * S + qi) * NH + h) * HD + hd] = oacc[c][r] / lrow[r];
    }
  }
}

// ---------------- router: probs = sigmoid(x . router_w) ----------------
__global__ __launch_bounds__(256) void router_kernel(
    const float* __restrict__ x, const float* __restrict__ rw,
    float* __restrict__ probs) {
  int t = blockIdx.x * 8 + (threadIdx.x >> 5);
  if (t >= T) return;
  int lane = threadIdx.x & 31;
  const float* xi = x + (size_t)t * D;
  float s = 0.f;
  for (int d = lane; d < D; d += 32) s += xi[d] * rw[d];
#pragma unroll
  for (int o = 16; o > 0; o >>= 1) s += __shfl_xor(s, o);
  if (lane == 0) probs[t] = 1.0f / (1.0f + __expf(-s));
}

// ---------------- per-batch top-k via LDS bitonic sort (descending) ----------
__global__ __launch_bounds__(1024) void topk_kernel(
    const float* __restrict__ probs, int* __restrict__ rowmap,
    float* __restrict__ rowprob) {
  __shared__ float sv[S];
  __shared__ int   si[S];
  const int b = blockIdx.x, tid = threadIdx.x;
  for (int i = tid; i < S; i += 1024) { sv[i] = probs[b * S + i]; si[i] = i; }
  __syncthreads();
  for (int k2 = 2; k2 <= S; k2 <<= 1) {
    for (int j = k2 >> 1; j > 0; j >>= 1) {
      for (int i = tid; i < S; i += 1024) {
        int ixj = i ^ j;
        if (ixj > i) {
          float a = sv[i], c = sv[ixj];
          int ia = si[i], ic = si[ixj];
          bool before = (a > c) || (a == c && ia < ic);
          bool desc = ((i & k2) == 0);
          if (desc ? !before : before) {
            sv[i] = c; sv[ixj] = a; si[i] = ic; si[ixj] = ia;
          }
        }
      }
      __syncthreads();
    }
  }
  for (int i = tid; i < KSEL; i += 1024) {
    rowmap[b * KSEL + i]  = b * S + si[i];
    rowprob[b * KSEL + i] = sv[i];
  }
}

// ---------------- host orchestration ----------------
extern "C" void kernel_launch(void* const* d_in, const int* in_sizes, int n_in,
                              void* d_out, int out_size, void* d_ws, size_t ws_size,
                              hipStream_t stream) {
  const int*   ids        = (const int*)  d_in[0];
  const int*   iteration  = (const int*)  d_in[1];
  const float* embedding  = (const float*)d_in[2];
  const float* iter_emb   = (const float*)d_in[3];
  const float* attn_norm  = (const float*)d_in[4];
  const float* Wqkv       = (const float*)d_in[5];
  const float* wo_w       = (const float*)d_in[6];
  const float* router_w   = (const float*)d_in[7];
  const float* mlp_norm   = (const float*)d_in[8];
  const float* gate_w     = (const float*)d_in[9];
  const float* up_w       = (const float*)d_in[10];
  const float* down_w     = (const float*)d_in[11];
  const float* final_norm = (const float*)d_in[12];
  float* out = (float*)d_out;

  float* ws = (float*)d_ws;
  size_t off = 0;
  auto alloc = [&](size_t n) { float* p = ws + off; off += n; return p; };
  float*    x       = alloc((size_t)T * D);
  float*    xn      = alloc((size_t)T * D);
  float*    qkv     = alloc((size_t)T * 3 * D);
  _Float16* qh      = (_Float16*)alloc((size_t)T * D / 2);
  _Float16* kh      = (_Float16*)alloc((size_t)T * D / 2);
  _Float16* vt      = (_Float16*)alloc((size_t)T * D / 2);
  float*    attno   = alloc((size_t)T * D);
  float*    probs   = alloc((size_t)T);
  float*    rowprob = alloc((size_t)B * KSEL);
  float*    selnorm = alloc((size_t)B * KSEL * D);
  float*    gatebuf = alloc((size_t)B * KSEL * FF);
  int*      rowmap  = (int*)(ws + off); off += (size_t)B * KSEL;

  const int TS = B * KSEL;   // 13104 selected tokens total

  embed_kernel<<<(T * D + 255) / 256, 256, 0, stream>>>(ids, iteration, embedding,
                                                        iter_emb, x);

  for (int l = 0; l < NL; ++l) {
    rmsnorm_kernel<<<(T + 7) / 8, 256, 0, stream>>>(x, attn_norm + (size_t)l * D,
                                                    xn, T, nullptr);
    gemm_kernel<<<dim3((3 * D + 127) / 128, (T + 63) / 64), 256, 0, stream>>>(
        xn, D, Wqkv + (size_t)l * 3 * D * D, D, qkv, 3 * D,
        T, 3 * D, D, 0, nullptr, nullptr);
    rope_kernel<<<(T * NH * HD + 255) / 256, 256, 0, stream>>>(qkv, qh, kh, vt);
    attn_kernel<<<dim3(S / 128, B * NH), 256, 0, stream>>>(qh, kh, vt, attno);
    gemm_kernel<<<dim3((D + 127) / 128, (T + 63) / 64), 256, 0, stream>>>(
        attno, D, wo_w + (size_t)l * D * D, D, x, D,
        T, D, D, 1, nullptr, nullptr);                       // residual add
    router_kernel<<<(T + 7) / 8, 256, 0, stream>>>(x, router_w + (size_t)l * D, probs);
    topk_kernel<<<B, 1024, 0, stream>>>(probs, rowmap, rowprob);
    rmsnorm_kernel<<<(TS + 7) / 8, 256, 0, stream>>>(x, mlp_norm + (size_t)l * D,
                                                     selnorm, TS, rowmap);
    gemm_kernel<<<dim3((FF + 127) / 128, (TS + 63) / 64), 256, 0, stream>>>(
        selnorm, D, gate_w + (size_t)l * FF * D, D, gatebuf, FF,
        TS, FF, D, 2, nullptr, nullptr);                     // silu(gate)
    gemm_kernel<<<dim3((FF + 127) / 128, (TS + 63) / 64), 256, 0, stream>>>(
        selnorm, D, up_w + (size_t)l * FF * D, D, gatebuf, FF,
        TS, FF, D, 3, nullptr, nullptr);                     // * up
    gemm_kernel<<<dim3((D + 127) / 128, (TS + 63) / 64), 256, 0, stream>>>(
        gatebuf, FF, down_w + (size_t)l * D * FF, FF, x, D,
        TS, D, FF, 4, rowmap, rowprob);                      // scatter-add*prob
  }
  rmsnorm_kernel<<<(T + 7) / 8, 256, 0, stream>>>(x, final_norm, out, T, nullptr);

  (void)in_sizes; (void)n_in; (void)out_size; (void)ws_size;
}
